// HebbianLearning_18691697672906
// MI455X (gfx1250) — matrix operand order repaired and checked by reading, not compiled, not run on hardware
//
#include <hip/hip_runtime.h>
#include <hip/hip_bf16.h>

typedef __attribute__((ext_vector_type(16))) __bf16       v16bf;
typedef __attribute__((ext_vector_type(8)))  float        v8f;
typedef __attribute__((ext_vector_type(8)))  unsigned int v8u;
typedef __attribute__((ext_vector_type(4)))  unsigned int v4u;
typedef __attribute__((ext_vector_type(8)))  int          v8i;
typedef __attribute__((ext_vector_type(4)))  int          v4i;

#define B_   32
#define C_   16
#define H_   64
#define W_   64
#define O_   256
#define KK_  5
#define HO_  60
#define WO_  60
#define NROWS (B_*HO_*WO_)        // 115200
#define D_    (C_*KK_*KK_)        // 400
#define DPAD  416                 // 13 * 32
#define YELEMS ((size_t)B_*O_*HO_*WO_)  // 29491200

static __device__ __forceinline__ unsigned short f2bf(float f) {
    unsigned u = __builtin_bit_cast(unsigned, f);
    unsigned r = u + 0x7FFFu + ((u >> 16) & 1u);
    return (unsigned short)(r >> 16);
}

// ---- CDNA5 async LDS->global data mover (ASYNCcnt-tracked) ----
static __device__ __forceinline__ void async_store_b128(unsigned lds_addr, void* gaddr) {
    asm volatile("global_store_async_from_lds_b128 %1, %0, off"
                 :: "v"(lds_addr), "v"((unsigned long long)(uintptr_t)gaddr)
                 : "memory");
}
static __device__ __forceinline__ void wait_asynccnt0() {
    asm volatile("s_wait_asynccnt 0x0" ::: "memory");
}

// ---------------------------------------------------------------------------
// Kernel 0: convert weights (padded to DPAD) and x to bf16 in workspace
// ---------------------------------------------------------------------------
__global__ void hebb_prep_kernel(const float* __restrict__ x,
                                 const float* __restrict__ w,
                                 unsigned short* __restrict__ wbf,
                                 unsigned short* __restrict__ xbf) {
    int id = blockIdx.x * blockDim.x + threadIdx.x;
    const int WTOT = O_ * DPAD;
    if (id < WTOT) {
        int o = id / DPAD, k = id - o * DPAD;
        unsigned short h = 0;
        if (k < D_) h = f2bf(w[o * D_ + k]);
        wbf[id] = h;
    } else {
        int id2 = id - WTOT;
        if (id2 < B_ * C_ * H_ * W_) xbf[id2] = f2bf(x[id2]);
    }
}

// ---------------------------------------------------------------------------
// Kernel 1: implicit-GEMM conv via v_wmma_f32_16x16x32_bf16
//   y[n, o] = sum_d x_unf[n, d] * w[o, d]
//   WG tile: 128 rows (n) x 64 channels (o); 8 waves, each 32x32 (2x2 WMMA)
//   B tile: TDM tensor_load_to_lds (one descriptor per k-step, wave 0).
//   Epilogue: async LDS->global streaming stores.
// ---------------------------------------------------------------------------
__global__ __launch_bounds__(256) void hebb_conv_wmma_kernel(
        const unsigned short* __restrict__ xbf,
        const unsigned short* __restrict__ wbf,
        float* __restrict__ y) {
    __shared__ unsigned short Alds[128 * 40];  // [m][k], stride 40 bf16
    __shared__ unsigned short Blds[64 * 40];   // [o][k], stride 40 bf16 (64B + 16B TDM pad)
    __shared__ float          Clds[64 * 132];  // [o][m], stride 132 f32 (16B aligned)

    const int tid  = threadIdx.x;
    const int o0   = blockIdx.x * 64;
    const int m0   = blockIdx.y * 128;

    const int lane = tid & 31;
    const int wave = tid >> 5;
    const int wm   = wave >> 1;   // 0..3 : 32-row band
    const int wn   = wave & 1;    // 0..1 : 32-channel band
    const int ll   = lane & 15;
    const int lh   = lane >> 4;

    const unsigned ldsB_base = (unsigned)(uintptr_t)&Blds[0];
    const unsigned ldsC_base = (unsigned)(uintptr_t)&Clds[0];

    // --- A im2col gather coordinates for this thread ---
    const int arow  = tid & 127;
    const int ahalf = tid >> 7;          // which 16-wide K half this thread fills
    const int gm    = m0 + arow;
    const int gb    = gm / 3600;
    const int grem  = gm - gb * 3600;
    const int goh   = grem / 60;
    const int gow   = grem - goh * 60;
    const unsigned short* xb = xbf + gb * (C_ * H_ * W_);

    // --- TDM descriptor for the weight tile: [tile 32(K) x 64(O)] of [416 x 256] bf16 ---
    // group1: data_size=2B (code 1), pad_enable, pad every 16 DWORDs (code 3) by
    //         4 DWORDs (code 3)  ->  LDS row stride = 64B data + 16B pad = 40 bf16
    const unsigned long long gW = (unsigned long long)(uintptr_t)(wbf + o0 * DPAD);
    v8i g1 = {};
    g1[0] = (1 << 16) | (1 << 20) | (3 << 22) | (3 << 25);          // data_size|pad_en|pad_int|pad_amt
    g1[1] = (int)(((unsigned)DPAD & 0xFFFFu) << 16);                // tensor_dim0[15:0]
    g1[2] = (int)(((unsigned)DPAD >> 16) | (((unsigned)O_ & 0xFFFFu) << 16)); // dim0 hi | dim1 lo
    g1[3] = (int)(((unsigned)O_ >> 16) | (32u << 16));              // dim1 hi | tile_dim0=32
    g1[4] = 64;                                                     // tile_dim1=64, tile_dim2=0
    g1[5] = DPAD;                                                   // tensor_dim0_stride[31:0]
    g1[6] = 0;
    g1[7] = 0;
    const v4i gz4 = {};
    const v8i gz8 = {};

    // --- B fallback coordinates reused for prefetch hints ---
    const int bnn = tid & 63;
    const int bch = tid >> 6;

    v8f acc[2][2] = {};

    for (int kt = 0; kt < DPAD / 32; ++kt) {
        const int k0 = kt * 32;

        // gather 16 bf16 A elements for (row=arow, k = k0 + ahalf*16 + j)
        unsigned int avals[8];
        #pragma unroll
        for (int j = 0; j < 16; ++j) {
            int k = k0 + ahalf * 16 + j;
            unsigned short h = 0;
            if (k < D_) {
                int c  = k / 25;
                int kr = k - c * 25;
                int r  = kr / 5;
                int s  = kr - r * 5;
                h = xb[(c << 12) + (goh + r) * 64 + (gow + s)];
            }
            if (j & 1) avals[j >> 1] |= ((unsigned)h) << 16;
            else       avals[j >> 1]  = (unsigned)h;
        }

        __syncthreads();  // previous tile fully consumed by all waves

        // wave 0 kicks the Tensor Data Mover for the weight tile
        if (wave == 0) {
            unsigned long long ga = gW + (unsigned)k0 * 2u;
            v4u g0;
            g0[0] = 1u;                                   // count=1 (valid user D#)
            g0[1] = ldsB_base;                            // LDS byte address
            g0[2] = (unsigned)ga;                         // global_addr[31:0]
            g0[3] = (unsigned)((ga >> 32) & 0x01FFFFFFull) | (2u << 30); // addr[56:32] | type=2
            __builtin_amdgcn_tensor_load_to_lds(g0, g1, gz4, gz4, gz8, 0);
        }
        if (kt + 1 < DPAD / 32)
            __builtin_prefetch(wbf + (o0 + bnn) * DPAD + k0 + 32 + bch * 8, 0, 1);

        *(uint4*)&Alds[arow * 40 + ahalf * 16]     = make_uint4(avals[0], avals[1], avals[2], avals[3]);
        *(uint4*)&Alds[arow * 40 + ahalf * 16 + 8] = make_uint4(avals[4], avals[5], avals[6], avals[7]);

        if (wave == 0) __builtin_amdgcn_s_wait_tensorcnt(0);  // TDM tile landed
        __syncthreads();                                      // A stores + B tile visible

        // --- load fragments per ISA 16-bit A (16x32) / B (32x16) layouts ---
        v16bf afrag[2], bfrag[2];
        #pragma unroll
        for (int mt = 0; mt < 2; ++mt) {
            int ar = wm * 32 + mt * 16 + ll;
            uint4 q0 = *(const uint4*)&Alds[ar * 40 + lh * 8];        // K = lh*8 .. +7
            uint4 q1 = *(const uint4*)&Alds[ar * 40 + 16 + lh * 8];   // K = 16+lh*8 .. +7
            v8u u;
            u[0] = q0.x; u[1] = q0.y; u[2] = q0.z; u[3] = q0.w;
            u[4] = q1.x; u[5] = q1.y; u[6] = q1.z; u[7] = q1.w;
            afrag[mt] = __builtin_bit_cast(v16bf, u);
        }
        #pragma unroll
        for (int nt = 0; nt < 2; ++nt) {
            int bn   = wn * 32 + nt * 16 + ll;
            int koff = lh * 8;                                        // K = koff..+7, 16+koff..+7
            uint4 p0 = *(const uint4*)&Blds[bn * 40 + koff];
            uint4 p1 = *(const uint4*)&Blds[bn * 40 + 16 + koff];
            v8u u;
            u[0] = p0.x; u[1] = p0.y; u[2] = p0.z; u[3] = p0.w;
            u[4] = p1.x; u[5] = p1.y; u[6] = p1.z; u[7] = p1.w;
            bfrag[nt] = __builtin_bit_cast(v16bf, u);
        }

        #pragma unroll
        for (int mt = 0; mt < 2; ++mt)
            #pragma unroll
            for (int nt = 0; nt < 2; ++nt)
                acc[mt][nt] = __builtin_amdgcn_wmma_f32_16x16x32_bf16(
                    false, afrag[mt], false, bfrag[nt],
                    (short)0, acc[mt][nt], false, false);
    }

    // --- epilogue: transpose through LDS, stream out with async LDS->global ---
    #pragma unroll
    for (int mt = 0; mt < 2; ++mt)
        #pragma unroll
        for (int nt = 0; nt < 2; ++nt) {
            int o_l = wn * 32 + nt * 16 + ll;
            int mb  = wm * 32 + mt * 16 + lh * 8;
            #pragma unroll
            for (int r = 0; r < 8; ++r)
                Clds[o_l * 132 + mb + r] = acc[mt][nt][r];
        }
    __syncthreads();

    // 64 channels x 128 rows = 2048 four-float chunks; each 16B, never crosses b
    for (int ch = tid; ch < 2048; ch += 256) {
        int m_c = (ch & 31) * 4;
        int o_l = ch >> 5;
        int n   = m0 + m_c;
        int b   = n / 3600;
        int rem = n - b * 3600;
        float* gp = y + (size_t)(b * O_ + o0 + o_l) * 3600 + rem;
        async_store_b128(ldsC_base + (unsigned)(o_l * 132 + m_c) * 4u, gp);
    }
    wait_asynccnt0();   // (S_ENDPGM would also wait-idle; explicit for clarity)
}

// ---------------------------------------------------------------------------
// Kernel 2: per-row max of s (row n = (b,oh,ow), columns o strided by 3600)
// ---------------------------------------------------------------------------
__global__ void hebb_rowmax_kernel(const float* __restrict__ y,
                                   float* __restrict__ rmax) {
    int n = blockIdx.x * blockDim.x + threadIdx.x;
    if (n >= NROWS) return;
    int b = n / 3600;
    int rem = n - b * 3600;
    const float* p = y + (size_t)b * O_ * 3600 + rem;
    float m = -__builtin_inff();
    for (int o = 0; o < O_; ++o) m = fmaxf(m, p[(size_t)o * 3600]);
    rmax[n] = m;
}

// ---------------------------------------------------------------------------
// Kernel 3: delta_w[o,:] = (1/N) sum_{n winner of o} v*x_unf[n,:] - mean(r_o)*w[o,:]
// One block per output channel; deterministic serial accumulation order.
// ---------------------------------------------------------------------------
__global__ __launch_bounds__(256) void hebb_delta_kernel(
        const float* __restrict__ x,
        const float* __restrict__ w,
        const float* __restrict__ y,
        const float* __restrict__ rmax,
        float* __restrict__ dw) {
    __shared__ float svals[256];
    __shared__ int   sflag[256];
    __shared__ float srsum;

    const int o = blockIdx.x;
    const int t = threadIdx.x;

    // per-thread patch coordinates for d0 = t and d1 = t + 256
    const int d0 = t, d1 = t + 256;
    const int c0 = d0 / 25;
    const int r0 = (d0 - c0 * 25) / 5;
    const int s0 = d0 - c0 * 25 - r0 * 5;
    const bool has1 = (d1 < D_);
    int c1 = 0, r1 = 0, s1 = 0;
    if (has1) { c1 = d1 / 25; int kr = d1 - c1 * 25; r1 = kr / 5; s1 = kr - r1 * 5; }

    float acc0 = 0.f, acc1 = 0.f, rs = 0.f;

    for (int base = 0; base < NROWS; base += 256) {
        const int n   = base + t;
        const int b   = n / 3600;
        const int rem = n - b * 3600;
        const float v = y[(size_t)b * (O_ * 3600) + (size_t)o * 3600 + rem];
        sflag[t] = (v == rmax[n]) ? 1 : 0;
        svals[t] = v;
        __syncthreads();

        for (int j = 0; j < 256; ++j) {
            if (sflag[j]) {
                const float vv  = svals[j];
                const int   nj  = base + j;
                const int   bj  = nj / 3600;
                const int   remj = nj - bj * 3600;
                const int   ohj = remj / 60;
                const int   owj = remj - ohj * 60;
                const float* xb = x + (size_t)bj * (C_ * H_ * W_);
                acc0 += vv * xb[(c0 << 12) + (ohj + r0) * 64 + owj + s0];
                if (has1)
                    acc1 += vv * xb[(c1 << 12) + (ohj + r1) * 64 + owj + s1];
                if (t == 0) rs += vv;
            }
        }
        __syncthreads();
    }

    if (t == 0) srsum = rs;
    __syncthreads();

    const float invN  = 1.0f / (float)NROWS;
    const float rmean = srsum * invN;
    dw[(size_t)o * D_ + d0] = acc0 * invN - rmean * w[(size_t)o * D_ + d0];
    if (has1)
        dw[(size_t)o * D_ + d1] = acc1 * invN - rmean * w[(size_t)o * D_ + d1];
}

// ---------------------------------------------------------------------------
extern "C" void kernel_launch(void* const* d_in, const int* in_sizes, int n_in,
                              void* d_out, int out_size, void* d_ws, size_t ws_size,
                              hipStream_t stream) {
    const float* x = (const float*)d_in[0];   // [32,16,64,64]
    const float* w = (const float*)d_in[1];   // [256,16,5,5]

    float* y  = (float*)d_out;                // [32,256,60,60]
    float* dw = y + YELEMS;                   // [256,16,5,5]

    unsigned short* wbf = (unsigned short*)d_ws;            // [256][416] bf16
    unsigned short* xbf = wbf + O_ * DPAD;                  // [32,16,64,64] bf16
    float*          rmx = (float*)(xbf + B_ * C_ * H_ * W_);// [115200] f32

    const int prep_total = O_ * DPAD + B_ * C_ * H_ * W_;
    hebb_prep_kernel<<<(prep_total + 255) / 256, 256, 0, stream>>>(x, w, wbf, xbf);

    hebb_conv_wmma_kernel<<<dim3(O_ / 64, NROWS / 128), 256, 0, stream>>>(xbf, wbf, y);

    hebb_rowmax_kernel<<<(NROWS + 255) / 256, 256, 0, stream>>>(y, rmx);

    hebb_delta_kernel<<<O_, 256, 0, stream>>>(x, w, y, rmx, dw);
}
